// Quantize_1726576854354
// MI455X (gfx1250) — matrix operand order repaired and checked by reading, not compiled, not run on hardware
//
#include <hip/hip_runtime.h>

#define NROWS   131072      // 32*64*64
#define DIM     256
#define NE      2048
#define QSIZE   (NROWS * DIM)   // 33554432

typedef __attribute__((ext_vector_type(8)))  float          v8f;
typedef __attribute__((ext_vector_type(16))) __bf16         v16bf;
typedef __attribute__((ext_vector_type(16))) unsigned short v16us;
typedef __attribute__((ext_vector_type(4)))  unsigned int   u32x4;
typedef __attribute__((ext_vector_type(4)))  int            i32x4;
typedef __attribute__((ext_vector_type(8)))  int            i32x8;

#if defined(__has_builtin)
#if __has_builtin(__builtin_amdgcn_tensor_load_to_lds) && __has_builtin(__builtin_amdgcn_s_wait_tensorcnt)
#define USE_TDM 1
#endif
#endif
#ifndef USE_TDM
#define USE_TDM 0
#endif

union Frag {
    uint4 q[2];
    v16us u;
    v16bf b;
};

__device__ __forceinline__ unsigned short f32_to_bf16_rn(float f) {
    unsigned u = __float_as_uint(f);
    u += 0x7fffu + ((u >> 16) & 1u);     // round-to-nearest-even
    return (unsigned short)(u >> 16);
}
__device__ __forceinline__ float bf16_to_f32(unsigned short h) {
    return __uint_as_float(((unsigned)h) << 16);
}

// ---------------------------------------------------------------------------
// Kernel 1: transpose codebook to [NE][DIM], split f32 -> bf16 hi/lo,
// compute ||e||^2 per code, zero the diff accumulator.
// ---------------------------------------------------------------------------
__global__ void vq_prep(const float* __restrict__ embed,      // [DIM][NE]
                        float* __restrict__ embT,             // [NE][DIM] f32
                        unsigned short* __restrict__ embT_hi, // [NE][DIM] bf16
                        unsigned short* __restrict__ embT_lo, // [NE][DIM] bf16
                        float* __restrict__ enorm,            // [NE]
                        float* __restrict__ diff_acc) {
    const int n = blockIdx.x * blockDim.x + threadIdx.x;
    if (n == 0) diff_acc[0] = 0.0f;
    if (n >= NE) return;
    float s = 0.0f;
    for (int k = 0; k < DIM; ++k) {
        float v = embed[(size_t)k * NE + n];   // coalesced across n
        s += v * v;
        embT[(size_t)n * DIM + k] = v;
        unsigned short h = f32_to_bf16_rn(v);
        embT_hi[(size_t)n * DIM + k] = h;
        embT_lo[(size_t)n * DIM + k] = f32_to_bf16_rn(v - bf16_to_f32(h));
    }
    enorm[n] = s;
}

#if USE_TDM
// Issue one TDM descriptor: a 2D tile, row0 = bf16-hi block, row1 = bf16-lo
// block (they sit exactly 1 MB = 262144 dwords apart in the workspace).
// Copies 32 codes x 256 K x (hi+lo) = 32 KB into LDS at lds_off.
__device__ __forceinline__ void tdm_issue(const unsigned short* gsrc, unsigned lds_off) {
    const unsigned long long ga = (unsigned long long)(size_t)gsrc;
    u32x4 g0 = {};
    g0[0] = 1u;                                   // count=1 (valid descriptor)
    g0[1] = lds_off;                              // LDS byte address
    g0[2] = (unsigned)ga;                         // global_addr[31:0]
    g0[3] = ((unsigned)(ga >> 32) & 0x01FFFFFFu)  // global_addr[56:32]
            | 0x80000000u;                        // type=2 ("image")
    i32x8 g1 = {};
    g1[0] = 0x00020000;   // workgroup_mask=0, data_size=2 (4 bytes)
    g1[1] = 0x10000000;   // tensor_dim0 = 4096 dwords (bits 79:48, low half)
    g1[2] = 0x00020000;   // tensor_dim0 hi = 0, tensor_dim1 = 2
    g1[3] = 0x10000000;   // tile_dim0 = 4096 dwords (bits 127:112)
    g1[4] = 0x00000002;   // tile_dim1 = 2, tile_dim2 = 0
    g1[5] = 0x00040000;   // tensor_dim0_stride = 262144 dwords (1 MB)
    g1[6] = 0;            // stride0 hi = 0, tensor_dim1_stride = 0
    g1[7] = 0;
    i32x4 z4 = {};
#if defined(__clang_major__) && (__clang_major__ >= 23)
    i32x8 z8 = {};
    __builtin_amdgcn_tensor_load_to_lds(g0, g1, z4, z4, z8, 0);
#else
    __builtin_amdgcn_tensor_load_to_lds(g0, g1, z4, z4, 0);
#endif
}
#endif

// ---------------------------------------------------------------------------
// Kernel 2: distance GEMM via bf16 WMMA (split-precision), fused argmin,
// codebook gather, quantize write, MSE partial reduction.
// Block: 256 threads = 8 waves; each wave owns 16 rows; block owns 128 rows.
// Each iteration handles 32 codes; B tiles double-buffered in LDS via TDM.
// ---------------------------------------------------------------------------
__global__ __launch_bounds__(256)
void vq_main(const float* __restrict__ X,               // [NROWS][DIM]
             const unsigned short* __restrict__ embT_hi,
             const unsigned short* __restrict__ embT_lo,
             const float* __restrict__ enorm,
             const float* __restrict__ embT,            // f32 [NE][DIM]
             float* __restrict__ out_q,                 // [NROWS][DIM]
             float* __restrict__ out_ind,               // [NROWS] (as float)
             float* __restrict__ diff_acc) {
    // per buffer: 32 codes * 256 K hi (16 KB) then lo (16 KB) = 32 KB
    __shared__ __align__(16) unsigned short s_b[2][16384];
    __shared__ int   s_idx[128];
    __shared__ float s_red[256];

    const int tid  = threadIdx.x;
    const int lane = tid & 31;
    const int wave = tid >> 5;             // 0..7
    const int hi16 = (lane >> 4) & 1;      // half-wave selector
    const int m    = lane & 15;            // row (A) / col (B) within 16-tile
    const int rowW = blockIdx.x * 128 + wave * 16;

    // ---- Load this wave's 16x256 A tile, split to bf16 hi/lo, keep in VGPRs.
    // ISA 16-bit A layout: lanes 0-15 hold K {c*32+0..7, c*32+16..23},
    //                      lanes 16-31 hold K {c*32+8..15, c*32+24..31}.
    v16bf ahi[8], alo[8];
    {
        const float* xr = X + (size_t)(rowW + m) * DIM;
        #pragma unroll
        for (int c = 0; c < 8; ++c) {
            const int kb = c * 32 + hi16 * 8;
            float4 f0 = *(const float4*)(xr + kb);
            float4 f1 = *(const float4*)(xr + kb + 4);
            float4 f2 = *(const float4*)(xr + kb + 16);
            float4 f3 = *(const float4*)(xr + kb + 20);
            float vals[16] = { f0.x,f0.y,f0.z,f0.w, f1.x,f1.y,f1.z,f1.w,
                               f2.x,f2.y,f2.z,f2.w, f3.x,f3.y,f3.z,f3.w };
            Frag h, l;
            #pragma unroll
            for (int e = 0; e < 16; ++e) {
                unsigned short hb = f32_to_bf16_rn(vals[e]);
                h.u[e] = hb;
                l.u[e] = f32_to_bf16_rn(vals[e] - bf16_to_f32(hb));
            }
            ahi[c] = h.b;
            alo[c] = l.b;
        }
    }

    float    best[8];
    unsigned bidx[8];
    #pragma unroll
    for (int r = 0; r < 8; ++r) { best[r] = __builtin_inff(); bidx[r] = 0u; }

#if USE_TDM
    const unsigned lds_off[2] = { (unsigned)(size_t)&s_b[0][0],
                                  (unsigned)(size_t)&s_b[1][0] };
    if (wave == 0) tdm_issue(embT_hi, lds_off[0]);    // prefetch tile 0
#endif

    // ---- Sweep all 2048 codes, 32 per iteration. Score = ||e||^2 - 2 x.e
    // (||x||^2 is constant per row -> cannot change the argmin).
    for (int t = 0; t < NE / 32; ++t) {
        const int colBase = t * 32;
        const int cur = t & 1;
#if USE_TDM
        if (wave == 0) __builtin_amdgcn_s_wait_tensorcnt(0);  // tile t landed
        __syncthreads();   // tile t visible to all; tile t-1 reads finished
        if (wave == 0 && t + 1 < NE / 32)                     // overlap t+1 DMA
            tdm_issue(embT_hi + (size_t)(colBase + 32) * DIM, lds_off[cur ^ 1]);
#else
        __syncthreads();   // LDS buffer reuse
        {   // cooperative stage: 32 codes x 256 K, hi then lo (32 KB)
            const uint4* gh = (const uint4*)(embT_hi + (size_t)colBase * DIM);
            const uint4* gl = (const uint4*)(embT_lo + (size_t)colBase * DIM);
            uint4* sh = (uint4*)&s_b[cur][0];
            uint4* sl = sh + 1024;
            #pragma unroll
            for (int i = 0; i < 4; ++i) {
                sh[tid + i * 256] = gh[tid + i * 256];
                sl[tid + i * 256] = gl[tid + i * 256];
            }
        }
        __syncthreads();
#endif

        v8f acc0 = {};
        v8f acc1 = {};
        #pragma unroll
        for (int c = 0; c < 8; ++c) {
            const int kb = c * 32 + hi16 * 8;
            const uint4* b0h = (const uint4*)(&s_b[cur][0] + m * DIM);
            const uint4* b1h = (const uint4*)(&s_b[cur][0] + (16 + m) * DIM);
            Frag f0h, f0l, f1h, f1l;
            f0h.q[0] = b0h[(kb >> 3)];
            f0h.q[1] = b0h[(kb >> 3) + 2];
            f0l.q[0] = b0h[(kb >> 3) + 1024];       // lo block: +8192 ushorts
            f0l.q[1] = b0h[(kb >> 3) + 1026];
            f1h.q[0] = b1h[(kb >> 3)];
            f1h.q[1] = b1h[(kb >> 3) + 2];
            f1l.q[0] = b1h[(kb >> 3) + 1024];
            f1l.q[1] = b1h[(kb >> 3) + 1026];
            // x.e ~= xhi.ehi + xhi.elo + xlo.ehi  (~2^-18 rel. error)
            acc0 = __builtin_amdgcn_wmma_f32_16x16x32_bf16(false, ahi[c], false, f0h.b, (short)0, acc0, false, false);
            acc1 = __builtin_amdgcn_wmma_f32_16x16x32_bf16(false, ahi[c], false, f1h.b, (short)0, acc1, false, false);
            acc0 = __builtin_amdgcn_wmma_f32_16x16x32_bf16(false, ahi[c], false, f0l.b, (short)0, acc0, false, false);
            acc1 = __builtin_amdgcn_wmma_f32_16x16x32_bf16(false, ahi[c], false, f1l.b, (short)0, acc1, false, false);
            acc0 = __builtin_amdgcn_wmma_f32_16x16x32_bf16(false, alo[c], false, f0h.b, (short)0, acc0, false, false);
            acc1 = __builtin_amdgcn_wmma_f32_16x16x32_bf16(false, alo[c], false, f1h.b, (short)0, acc1, false, false);
        }

        const unsigned n0  = (unsigned)(colBase + m);
        const unsigned n1  = (unsigned)(colBase + 16 + m);
        const float    en0 = enorm[n0];
        const float    en1 = enorm[n1];
        #pragma unroll
        for (int r = 0; r < 8; ++r) {   // C layout: VGPR r -> row r + 8*hi16
            float d0 = en0 - 2.0f * acc0[r];
            if (d0 < best[r] || (d0 == best[r] && n0 < bidx[r])) { best[r] = d0; bidx[r] = n0; }
            float d1 = en1 - 2.0f * acc1[r];
            if (d1 < best[r] || (d1 == best[r] && n1 < bidx[r])) { best[r] = d1; bidx[r] = n1; }
        }
    }

    // ---- argmin across the 16 lanes of each half-wave (rows differ per half)
    #pragma unroll
    for (int r = 0; r < 8; ++r) {
        float    bv = best[r];
        unsigned bi = bidx[r];
        #pragma unroll
        for (int mask = 8; mask >= 1; mask >>= 1) {
            float    ov = __shfl_xor(bv, mask, 32);
            unsigned oi = __shfl_xor(bi, mask, 32);
            if (ov < bv || (ov == bv && oi < bi)) { bv = ov; bi = oi; }
        }
        if (m == 0)   // lane 0 -> rows 0..7, lane 16 -> rows 8..15
            s_idx[wave * 16 + hi16 * 8 + r] = (int)bi;
    }
    __syncthreads();

    // ---- write indices, gather exact f32 codes, accumulate MSE
    const int rowBase = blockIdx.x * 128;
    if (tid < 128) out_ind[rowBase + tid] = (float)s_idx[tid];

    float dsum = 0.0f;
    for (int r = 0; r < 128; ++r) {
        const size_t row = (size_t)(rowBase + r);
        const float q = embT[(size_t)s_idx[r] * DIM + tid];
        const float x = X[row * DIM + tid];
        out_q[row * DIM + tid] = q;
        const float e = q - x;
        dsum += e * e;
    }
    s_red[tid] = dsum;
    __syncthreads();
    for (int s = 128; s > 0; s >>= 1) {
        if (tid < s) s_red[tid] += s_red[tid + s];
        __syncthreads();
    }
    if (tid == 0) atomicAdd(diff_acc, s_red[0]);
}

// ---------------------------------------------------------------------------
// Kernel 3: finalize mean
// ---------------------------------------------------------------------------
__global__ void vq_finalize(const float* __restrict__ diff_acc,
                            float* __restrict__ out_diff) {
    out_diff[0] = diff_acc[0] * (1.0f / (float)QSIZE);
}

extern "C" void kernel_launch(void* const* d_in, const int* in_sizes, int n_in,
                              void* d_out, int out_size, void* d_ws, size_t ws_size,
                              hipStream_t stream) {
    const float* X     = (const float*)d_in[0];   // [32,64,64,256]
    const float* embed = (const float*)d_in[1];   // [256,2048]

    float* out      = (float*)d_out;
    float* out_q    = out;                            // 33554432 floats
    float* out_diff = out + (size_t)QSIZE;            // 1 float
    float* out_ind  = out + (size_t)QSIZE + 1;        // 131072 floats

    char* ws = (char*)d_ws;
    float*          embT    = (float*)ws;                                   // 2 MB
    unsigned short* embT_hi = (unsigned short*)(ws + (size_t)NE * DIM * 4); // 1 MB
    unsigned short* embT_lo = (unsigned short*)(ws + (size_t)NE * DIM * 6); // 1 MB (exactly 1 MB after embT_hi)
    float*          enorm   = (float*)(ws + (size_t)NE * DIM * 8);          // 8 KB
    float*          diffacc = (float*)(ws + (size_t)NE * DIM * 8 + NE * 4); // 4 B

    vq_prep<<<NE / 256, 256, 0, stream>>>(embed, embT, embT_hi, embT_lo, enorm, diffacc);
    vq_main<<<NROWS / 128, 256, 0, stream>>>(X, embT_hi, embT_lo, enorm, embT,
                                             out_q, out_ind, diffacc);
    vq_finalize<<<1, 1, 0, stream>>>(diffacc, out_diff);
}
